// ScaledDotProductAttentionMemory_21371757265613
// MI455X (gfx1250) — matrix operand order repaired and checked by reading, not compiled, not run on hardware
//
#include <hip/hip_runtime.h>
#include <hip/hip_bf16.h>

// ---------------------------------------------------------------------------
// Problem constants (from the reference)
// ---------------------------------------------------------------------------
constexpr int B_    = 4;
constexpr int NQ_   = 2048;
constexpr int NK_   = 2048;
constexpr int H_    = 8;
constexpr int DK_   = 64;
constexpr int DV_   = 64;
constexpr int M_    = 40;
constexpr int DM_   = 512;          // d_model = H*DK
constexpr int NKM_  = NK_ + M_;     // 2088 valid key/value rows
constexpr int NKP_  = 2112;         // padded to 66 tiles of 32
constexpr float INV_SQRT_DK = 0.125f;      // 1/sqrt(64)
constexpr float SQRT_DK     = 8.0f;        // sqrt(64)
constexpr float SQRT_M      = 6.32455532f; // sqrt(40)
constexpr float NEG_BIG     = -1e30f;      // finite stand-in for -inf

// ---------------------------------------------------------------------------
// WMMA types / helpers (CDNA5 gfx1250, wave32)
// ---------------------------------------------------------------------------
typedef __attribute__((ext_vector_type(16))) __bf16 v16bf;
typedef __attribute__((ext_vector_type(8)))  float  v8f;
typedef __attribute__((ext_vector_type(4)))  int    v4i;

#define AS1 __attribute__((address_space(1)))
#define AS3 __attribute__((address_space(3)))

union V16U {
    v16bf  v;
    uint4  u4[2];
    __bf16 e[16];
};

__device__ __forceinline__ v8f wmma_bf16(v16bf a, v16bf b, v8f c) {
    // D = A(16x32 bf16) * B(32x16 bf16) + C(16x16 f32)
    return __builtin_amdgcn_wmma_f32_16x16x32_bf16(
        false, a, false, b, (short)0, c, false, false);
}

// Pack two f32 into one dword of two bf16 (lowers to v_cvt_pk_bf16_f32).
__device__ __forceinline__ unsigned pack_bf16(float a, float b) {
    union { __bf16 h[2]; unsigned u; } r;
    r.h[0] = (__bf16)a; r.h[1] = (__bf16)b;
    return r.u;
}

// A fragment (16x32, MxK). Layout: lane<16 -> M=lane, K 0..15;
// lane>=16 -> M=lane-16, K 16..31. Two 16-byte contiguous loads.
__device__ __forceinline__ v16bf load_a_frag(const __bf16* base, int ld) {
    const int lane = threadIdx.x & 31;
    const int row  = lane & 15;
    const int kb   = (lane >> 4) << 4;       // 0 or 16
    const __bf16* p = base + row * ld + kb;
    V16U r;
    r.u4[0] = *(const uint4*)(p);
    r.u4[1] = *(const uint4*)(p + 8);
    return r.v;
}

// B fragment (32x16, KxN) from a tile stored TRANSPOSED as [N][K] row-major
// (one 32-byte contiguous read per lane -> two ds_load_b128).
__device__ __forceinline__ v16bf load_b_frag_T(const __bf16* base, int ld) {
    const int lane = threadIdx.x & 31;
    const int n  = lane & 15;
    const int kb = (lane >> 4) << 4;         // 0 or 16
    const __bf16* p = base + n * ld + kb;
    V16U r;
    r.u4[0] = *(const uint4*)(p);
    r.u4[1] = *(const uint4*)(p + 8);
    return r.v;
}

// B fragment (32x16, KxN) from a tile stored NATURALLY as [K][N] row-major
// (16-bit gathers; only used in the projection GEMMs, not in hot loops).
__device__ __forceinline__ v16bf load_b_frag_N(const __bf16* base, int ld) {
    const int lane = threadIdx.x & 31;
    const int n  = lane & 15;
    const int kb = (lane >> 4) << 4;
    V16U r;
#pragma unroll
    for (int j = 0; j < 16; ++j) r.e[j] = base[(kb + j) * ld + n];
    return r.v;
}

// ---------------------------------------------------------------------------
// gfx1250 async global->LDS copy (ASYNCcnt-tracked, bypasses VGPRs).
// Builtin prototype: int4-addrspace(1) src, int4-addrspace(3) dst, imm, imm.
// ---------------------------------------------------------------------------
__device__ __forceinline__ void async_b128(__bf16* lds_dst, const __bf16* gsrc) {
#if __has_builtin(__builtin_amdgcn_global_load_async_to_lds_b128)
    void* s = (void*)gsrc;
    void* d = (void*)lds_dst;
    __builtin_amdgcn_global_load_async_to_lds_b128(
        (AS1 v4i*)s, (AS3 v4i*)d, 0, 0);
#else
    const unsigned loff = (unsigned)(unsigned long long)lds_dst;
    asm volatile("global_load_async_to_lds_b128 %0, %1, off"
                 :: "v"(loff), "v"(gsrc) : "memory");
#endif
}

__device__ __forceinline__ void wait_async0() {
#if __has_builtin(__builtin_amdgcn_s_wait_asynccnt)
    __builtin_amdgcn_s_wait_asynccnt(0);
#else
    asm volatile("s_wait_asynccnt 0" ::: "memory");
#endif
}

// ---------------------------------------------------------------------------
// Kernel 1: fp32 GEMM (8192x512) @ (512x512) + bias -> bf16.
// VTRANS=false: scatter to (b, h, seq, 64) with sequence stride Sdim.
// VTRANS=true  (V projection): scatter to transposed (b, h, dv, key) with key
// stride NKP_ — each lane's 8 accumulator rows are consecutive keys, packed
// into a single b128 store.
// ---------------------------------------------------------------------------
template <bool VTRANS>
__global__ __launch_bounds__(256)
void proj_gemm_kernel(const float* __restrict__ X, const float* __restrict__ W,
                      const float* __restrict__ bias, __bf16* __restrict__ Out,
                      int Sdim)
{
    constexpr int ALD = 40;                  // padded LDS strides (16B aligned)
    constexpr int WLD = 72;
    __shared__ __bf16 Al[128 * ALD];         // A tile 128x32 bf16, [m][k]
    __shared__ __bf16 Wl[32 * WLD];          // W tile 32x64 bf16, natural [k][n]

    const int tid  = threadIdx.x;
    const int lane = tid & 31;
    const int wv   = tid >> 5;               // wave id 0..7 -> 16-row slice
    const int r0   = blockIdx.y * 128;
    const int c0   = blockIdx.x * 64;
    const int grp  = lane >> 4;
    const int ncol = lane & 15;

    v8f acc[4] = {};

    for (int k0 = 0; k0 < DM_; k0 += 32) {
        // Stage A (fp32 -> bf16): packed pairs -> single b128 store per chunk.
#pragma unroll
        for (int i = 0; i < 2; ++i) {
            const int c   = tid + i * 256;
            const int row = c >> 2;
            const int seg = c & 3;
            const float* sp = X + (size_t)(r0 + row) * DM_ + k0 + seg * 8;
            const float4 f0 = *(const float4*)(sp);
            const float4 f1 = *(const float4*)(sp + 4);
            uint4 p;
            p.x = pack_bf16(f0.x, f0.y); p.y = pack_bf16(f0.z, f0.w);
            p.z = pack_bf16(f1.x, f1.y); p.w = pack_bf16(f1.z, f1.w);
            *(uint4*)&Al[row * ALD + seg * 8] = p;
        }
        // Stage W naturally ([k][n], coalesced float4 reads): 1 chunk/thread.
        {
            const int k   = tid >> 3;        // 0..31
            const int seg = tid & 7;         // n = seg*8 .. seg*8+7
            const float* sp = W + (size_t)(k0 + k) * DM_ + c0 + seg * 8;
            const float4 f0 = *(const float4*)(sp);
            const float4 f1 = *(const float4*)(sp + 4);
            uint4 p;
            p.x = pack_bf16(f0.x, f0.y); p.y = pack_bf16(f0.z, f0.w);
            p.z = pack_bf16(f1.x, f1.y); p.w = pack_bf16(f1.z, f1.w);
            *(uint4*)&Wl[k * WLD + seg * 8] = p;
        }
        __syncthreads();

        const v16bf a = load_a_frag(&Al[(wv * 16) * ALD], ALD);
#pragma unroll
        for (int nt = 0; nt < 4; ++nt) {
            const v16bf b = load_b_frag_N(&Wl[nt * 16], WLD);
            acc[nt] = wmma_bf16(a, b, acc[nt]);
        }
        __syncthreads();
    }

    // Epilogue: + bias, scatter
#pragma unroll
    for (int nt = 0; nt < 4; ++nt) {
        const int col = c0 + nt * 16 + ncol;
        const float bs = bias[col];
        const int h  = col >> 6;
        const int dk = col & 63;
        if (VTRANS) {
            const int gr0   = r0 + wv * 16 + grp * 8;   // 8 consecutive rows (keys)
            const int b     = gr0 >> 11;
            const int srow0 = gr0 & 2047;
            uint4 p;
            p.x = pack_bf16(acc[nt][0] + bs, acc[nt][1] + bs);
            p.y = pack_bf16(acc[nt][2] + bs, acc[nt][3] + bs);
            p.z = pack_bf16(acc[nt][4] + bs, acc[nt][5] + bs);
            p.w = pack_bf16(acc[nt][6] + bs, acc[nt][7] + bs);
            *(uint4*)&Out[((size_t)((b * H_ + h) * DV_ + dk)) * NKP_ + srow0] = p;
        } else {
#pragma unroll
            for (int j = 0; j < 8; ++j) {
                const int gr   = r0 + wv * 16 + grp * 8 + j;
                const int b    = gr >> 11;
                const int srow = gr & 2047;
                Out[(((size_t)(b * H_ + h) * Sdim + srow) * DK_) + dk] =
                    (__bf16)(acc[nt][j] + bs);
            }
        }
    }
}

// ---------------------------------------------------------------------------
// Kernel 2: append scaled memory slots (rows NK..NK+M-1) and zero-pad to NKP.
// K natural layout; V transposed layout.
// ---------------------------------------------------------------------------
__global__ __launch_bounds__(256)
void memslot_kernel(const float* __restrict__ mk, const float* __restrict__ mv,
                    __bf16* __restrict__ Kb, __bf16* __restrict__ Vt)
{
    const int idx = blockIdx.x * blockDim.x + threadIdx.x;
    if (idx >= B_ * H_ * (NKP_ - NK_) * DK_) return;
    const int dk = idx & 63;
    const int mi = (idx >> 6) & 63;
    const int h  = (idx >> 12) & 7;
    const int b  = idx >> 15;
    float kv = 0.f, vv = 0.f;
    if (mi < M_) {
        kv = SQRT_DK * mk[(size_t)mi * DM_ + h * DK_ + dk];
        vv = SQRT_M  * mv[(size_t)mi * DM_ + h * DK_ + dk];
    }
    Kb[((size_t)(b * H_ + h) * NKP_ + NK_ + mi) * DK_ + dk] = (__bf16)kv;
    Vt[((size_t)((b * H_ + h) * DV_ + dk)) * NKP_ + NK_ + mi] = (__bf16)vv;
}

// ---------------------------------------------------------------------------
// Kernel 3: fused flash attention with memory slots.
// Block = 256 threads = 8 waves; each wave owns a 16-query tile. K tiles
// ([32 keys][64 dk]) and transposed V tiles ([64 dv][32 keys]) are
// double-buffered in LDS via gfx1250 async global->LDS copies: issue tile
// i+1, compute on tile i, s_wait_asynccnt 0, one barrier per iteration.
// All WMMA operand reads are contiguous ds_load_b128.
// ---------------------------------------------------------------------------
__global__ __launch_bounds__(256)
void attn_kernel(const __bf16* __restrict__ Qb, const __bf16* __restrict__ Kb,
                 const __bf16* __restrict__ Vt, const float* __restrict__ AW,
                 const unsigned char* __restrict__ AM, __bf16* __restrict__ Xb)
{
    constexpr int KLD = 72;                  // K tile stride (64 dk + pad)
    constexpr int VLD = 40;                  // V tile stride (32 keys + pad)
    constexpr int PLD = 40;                  // P tile stride
    constexpr int NT  = NKP_ / 32;           // 66 key tiles
    __shared__ __bf16 Kl[2][32 * KLD];       // [key][dk]
    __shared__ __bf16 Vl[2][64 * VLD];       // [dv][key]  (transposed)
    __shared__ __bf16 Pl[8][16 * PLD];       // per-wave P staging (C/D -> A frag)

    const int tid  = threadIdx.x;
    const int lane = tid & 31;
    const int wv   = tid >> 5;
    const int bh   = blockIdx.y;             // b*H + h
    const int b    = bh >> 3;
    const int h    = bh & 7;
    const int qbase = blockIdx.x * 128 + wv * 16;
    const int grp  = lane >> 4;
    const int ncol = lane & 15;

    // This thread's slice of the cooperative tile copy (32B = 2 async b128).
    // tid 0..127   -> K tile row kr (0..31), 32B segment ks (0..3)
    // tid 128..255 -> V tile row vr (0..63, dv), 32B segment vs (0..1)
    const bool sIsK = tid < 128;
    const int kr = (tid & 127) >> 2, ks = tid & 3;
    const int vr = (tid & 127) >> 1, vs = tid & 1;
    const __bf16* sbase = sIsK
        ? Kb + ((size_t)bh * NKP_ + kr) * DK_ + ks * 16
        : Vt + ((size_t)(bh * DV_ + vr)) * NKP_ + vs * 16;
    const int    ldoff   = sIsK ? (kr * KLD + ks * 16) : (vr * VLD + vs * 16);
    const size_t sstride = sIsK ? (size_t)DK_ : (size_t)1;   // elems per key step

    // Q fragments for this 16-query tile (loaded once, kept in VGPRs)
    const __bf16* qptr = Qb + ((size_t)bh * NQ_ + qbase) * DK_;
    const v16bf aq0 = load_a_frag(qptr, DK_);        // dk 0..31
    const v16bf aq1 = load_a_frag(qptr + 32, DK_);   // dk 32..63

    float rm[8], rs[8];
#pragma unroll
    for (int j = 0; j < 8; ++j) { rm[j] = NEG_BIG; rs[j] = 0.f; }
    v8f acc[4] = {};

    // Prologue: async-stage tile 0 into buffer 0
    {
        __bf16* dst = (sIsK ? &Kl[0][0] : &Vl[0][0]) + ldoff;
        async_b128(dst, sbase);
        async_b128(dst + 8, sbase + 8);
    }
    wait_async0();
    __syncthreads();

    for (int it = 0; it < NT; ++it) {
        const int k0  = it * 32;
        const int buf = it & 1;

        // Prefetch next tile into the other buffer (overlaps with compute)
        if (it + 1 < NT) {
            const __bf16* src = sbase + (size_t)(k0 + 32) * sstride;
            __bf16* dst = (sIsK ? &Kl[buf ^ 1][0] : &Vl[buf ^ 1][0]) + ldoff;
            async_b128(dst, src);
            async_b128(dst + 8, src + 8);
        }

        const __bf16* kl = &Kl[buf][0];
        const __bf16* vl = &Vl[buf][0];

        // S = Q K^T for keys [k0, k0+32): two 16x16 f32 tiles, 4 WMMAs
        v8f s0 = {}, s1 = {};
        s0 = wmma_bf16(aq0, load_b_frag_T(kl, KLD), s0);
        s0 = wmma_bf16(aq1, load_b_frag_T(kl + 32, KLD), s0);
        s1 = wmma_bf16(aq0, load_b_frag_T(kl + 16 * KLD, KLD), s1);
        s1 = wmma_bf16(aq1, load_b_frag_T(kl + 16 * KLD + 32, KLD), s1);

        // Scale + attention_weights multiply + mask + pad handling
        const int kg0 = k0 + ncol;
        const int kg1 = k0 + 16 + ncol;
        float p0a[8], p1a[8];
#pragma unroll
        for (int j = 0; j < 8; ++j) {
            const int q = qbase + grp * 8 + j;
            const size_t rowoff = ((size_t)bh * NQ_ + q) * NK_;
            float v0 = s0[j] * INV_SQRT_DK;
            float v1 = s1[j] * INV_SQRT_DK;
            if (kg0 < NK_) {
                v0 = AM[rowoff + kg0] ? NEG_BIG : v0 * AW[rowoff + kg0];
            } else if (kg0 >= NKM_) v0 = NEG_BIG;
            if (kg1 < NK_) {
                v1 = AM[rowoff + kg1] ? NEG_BIG : v1 * AW[rowoff + kg1];
            } else if (kg1 >= NKM_) v1 = NEG_BIG;
            p0a[j] = v0; p1a[j] = v1;
        }

        // Online softmax per query row (rows live across 16-lane groups)
#pragma unroll
        for (int j = 0; j < 8; ++j) {
            float m = fmaxf(p0a[j], p1a[j]);
#pragma unroll
            for (int xm = 8; xm >= 1; xm >>= 1) m = fmaxf(m, __shfl_xor(m, xm, 16));
            const float mnew  = fmaxf(rm[j], m);
            const float scale = __expf(rm[j] - mnew);
            rm[j] = mnew;
            const float e0 = __expf(p0a[j] - mnew);
            const float e1 = __expf(p1a[j] - mnew);
            float rsum = e0 + e1;
#pragma unroll
            for (int xm = 8; xm >= 1; xm >>= 1) rsum += __shfl_xor(rsum, xm, 16);
            rs[j] = rs[j] * scale + rsum;
            p0a[j] = e0; p1a[j] = e1;
#pragma unroll
            for (int nt = 0; nt < 4; ++nt) acc[nt][j] *= scale;
        }

        // Re-layout P (C/D frag) -> A frag via per-wave LDS, then P @ V.
        // V tile is transposed in LDS, so B fragments are contiguous b128 loads.
        __bf16* pl = &Pl[wv][0];
#pragma unroll
        for (int j = 0; j < 8; ++j) {
            pl[(grp * 8 + j) * PLD + ncol]      = (__bf16)p0a[j];
            pl[(grp * 8 + j) * PLD + 16 + ncol] = (__bf16)p1a[j];
        }
        const v16bf ap = load_a_frag(pl, PLD);
#pragma unroll
        for (int nt = 0; nt < 4; ++nt) {
            const v16bf bvf = load_b_frag_T(vl + (nt * 16) * VLD, VLD);
            acc[nt] = wmma_bf16(ap, bvf, acc[nt]);
        }

        // My prefetch is done; barrier makes everyone's prefetch + reads done
        wait_async0();
        __syncthreads();
    }

    // Normalize and write bf16 attention output as (b, q, h*DV + dv)
#pragma unroll
    for (int nt = 0; nt < 4; ++nt) {
#pragma unroll
        for (int j = 0; j < 8; ++j) {
            const int q = qbase + grp * 8 + j;
            const float o = acc[nt][j] / rs[j];
            Xb[((size_t)(b * NQ_ + q)) * DM_ + h * DV_ + nt * 16 + ncol] = (__bf16)o;
        }
    }
}

// ---------------------------------------------------------------------------
// Kernel 4: output projection, bf16 X (8192x512) @ fp32 Wo (512x512) + bo -> f32
// ---------------------------------------------------------------------------
__global__ __launch_bounds__(256)
void out_gemm_kernel(const __bf16* __restrict__ X, const float* __restrict__ W,
                     const float* __restrict__ bias, float* __restrict__ Out)
{
    constexpr int ALD = 40;
    constexpr int WLD = 72;
    __shared__ __bf16 Al[128 * ALD];
    __shared__ __bf16 Wl[32 * WLD];

    const int tid  = threadIdx.x;
    const int lane = tid & 31;
    const int wv   = tid >> 5;
    const int r0   = blockIdx.y * 128;
    const int c0   = blockIdx.x * 64;
    const int grp  = lane >> 4;
    const int ncol = lane & 15;

    v8f acc[4] = {};

    for (int k0 = 0; k0 < DM_; k0 += 32) {
        // Stage A (already bf16): 512 16B chunks, 2 per thread
#pragma unroll
        for (int i = 0; i < 2; ++i) {
            const int c   = tid + i * 256;
            const int row = c >> 2;
            const int seg = c & 3;
            *(uint4*)&Al[row * ALD + seg * 8] =
                *(const uint4*)(X + (size_t)(r0 + row) * DM_ + k0 + seg * 8);
        }
        // Stage W naturally
        {
            const int k   = tid >> 3;
            const int seg = tid & 7;
            const float* sp = W + (size_t)(k0 + k) * DM_ + c0 + seg * 8;
            const float4 f0 = *(const float4*)(sp);
            const float4 f1 = *(const float4*)(sp + 4);
            uint4 p;
            p.x = pack_bf16(f0.x, f0.y); p.y = pack_bf16(f0.z, f0.w);
            p.z = pack_bf16(f1.x, f1.y); p.w = pack_bf16(f1.z, f1.w);
            *(uint4*)&Wl[k * WLD + seg * 8] = p;
        }
        __syncthreads();

        const v16bf a = load_a_frag(&Al[(wv * 16) * ALD], ALD);
#pragma unroll
        for (int nt = 0; nt < 4; ++nt) {
            const v16bf b = load_b_frag_N(&Wl[nt * 16], WLD);
            acc[nt] = wmma_bf16(a, b, acc[nt]);
        }
        __syncthreads();
    }

#pragma unroll
    for (int nt = 0; nt < 4; ++nt) {
        const int col = c0 + nt * 16 + ncol;
        const float bs = bias[col];
#pragma unroll
        for (int j = 0; j < 8; ++j) {
            const int gr = r0 + wv * 16 + grp * 8 + j;
            Out[(size_t)gr * DM_ + col] = acc[nt][j] + bs;
        }
    }
}

// ---------------------------------------------------------------------------
// Host launch. Workspace layout (bf16 elements), ~34.1 MB total:
//   Qb: B*H*NQ*DK | Kb: B*H*NKP*DK | Vt: B*H*DV*NKP (transposed) | Xb: B*NQ*DM
// ---------------------------------------------------------------------------
extern "C" void kernel_launch(void* const* d_in, const int* in_sizes, int n_in,
                              void* d_out, int out_size, void* d_ws, size_t ws_size,
                              hipStream_t stream) {
    const float* queries = (const float*)d_in[0];
    const float* keys    = (const float*)d_in[1];
    const float* values  = (const float*)d_in[2];
    const float* aw      = (const float*)d_in[3];
    const unsigned char* am = (const unsigned char*)d_in[4];  // jax bool -> 1 byte
    const float* Wq = (const float*)d_in[5];
    const float* bq = (const float*)d_in[6];
    const float* Wk = (const float*)d_in[7];
    const float* bk = (const float*)d_in[8];
    const float* Wv = (const float*)d_in[9];
    const float* bv = (const float*)d_in[10];
    const float* Wo = (const float*)d_in[11];
    const float* bo = (const float*)d_in[12];
    const float* mk = (const float*)d_in[13];
    const float* mv = (const float*)d_in[14];
    float* out = (float*)d_out;
    (void)in_sizes; (void)n_in; (void)out_size; (void)ws_size;

    __bf16* Qb = (__bf16*)d_ws;
    __bf16* Kb = Qb + (size_t)B_ * H_ * NQ_ * DK_;
    __bf16* Vt = Kb + (size_t)B_ * H_ * NKP_ * DK_;
    __bf16* Xb = Vt + (size_t)B_ * H_ * DV_ * NKP_;

    const dim3 blk(256);
    const dim3 gproj(DM_ / 64, (B_ * NQ_) / 128);     // (8, 64)

    proj_gemm_kernel<false><<<gproj, blk, 0, stream>>>(queries, Wq, bq, Qb, NQ_);
    proj_gemm_kernel<false><<<gproj, blk, 0, stream>>>(keys,    Wk, bk, Kb, NKP_);
    proj_gemm_kernel<true ><<<gproj, blk, 0, stream>>>(values,  Wv, bv, Vt, NKP_);

    const int mthreads = B_ * H_ * (NKP_ - NK_) * DK_;
    memslot_kernel<<<(mthreads + 255) / 256, blk, 0, stream>>>(mk, mv, Kb, Vt);

    attn_kernel<<<dim3(NQ_ / 128, B_ * H_), blk, 0, stream>>>(Qb, Kb, Vt, aw, am, Xb);

    out_gemm_kernel<<<gproj, blk, 0, stream>>>(Xb, Wo, bo, out);
}
